// SenseEmbedding_40295383171457
// MI455X (gfx1250) — matrix operand order repaired and checked by reading, compiled, not run on hardware
//
#include <hip/hip_runtime.h>
#include <cstdint>

#define NUM_SENSES 8
#define VEC_DIM    128
#define NCTX       10
#define NIDX       12
#define WAVES_PER_BLOCK 8
#define THREADS   (WAVES_PER_BLOCK * 32)
#define ROWB      544                       /* 512B sense row + 32B bank pad  */
#define CTX_OFF   (NUM_SENSES * ROWB)       /* 4352                            */
#define WAVE_LDS  (CTX_OFF + VEC_DIM * 4)   /* 4864 bytes per wave             */

typedef __attribute__((ext_vector_type(16))) _Float16 v16h;
typedef __attribute__((ext_vector_type(8)))  float    v8f;

__global__ __launch_bounds__(THREADS)
void SenseEmbedding_kernel(const int* __restrict__ x,
                           const float* __restrict__ Wg,
                           const float* __restrict__ Ws,
                           float* __restrict__ out,
                           int nrows)
{
    __shared__ __align__(16) unsigned char lds_raw[WAVES_PER_BLOCK * WAVE_LDS];

    const int lane = threadIdx.x & 31;
    const int wid  = (int)blockIdx.x * WAVES_PER_BLOCK + (int)(threadIdx.x >> 5);
    if (wid >= nrows) return;                       // wave-uniform guard

    unsigned char* base  = lds_raw + (size_t)(threadIdx.x >> 5) * WAVE_LDS;
    const uint32_t lbase = (uint32_t)(uintptr_t)base;   // block-relative LDS byte addr

    const int* xb = x + (size_t)wid * NIDX;
    const int  x0 = xb[0];
    const int  x1 = xb[1];

    // ---- Async copy W_s[x0] (8 rows x 512B) into LDS; overlaps context gathers.
    {
        const uint64_t gs =
            (uint64_t)(uintptr_t)(Ws + (size_t)x0 * (NUM_SENSES * VEC_DIM));
#pragma unroll
        for (int r = 0; r < NUM_SENSES; ++r) {
            uint32_t laddr = lbase + (uint32_t)(r * ROWB + lane * 16);
            uint64_t gaddr = gs + (uint64_t)(r * 512 + lane * 16);
            asm volatile("global_load_async_to_lds_b128 %0, %1, off"
                         :: "v"(laddr), "v"(gaddr) : "memory");
        }
    }

    // ---- Target row (coalesced b128 per lane), issued early.
    const float4 t = ((const float4*)(Wg + (size_t)x1 * VEC_DIM))[lane];

    // ---- Context sum: 10 coalesced row gathers, lane owns dims [4*lane, 4*lane+4).
    float4 acc = make_float4(0.f, 0.f, 0.f, 0.f);
#pragma unroll
    for (int j = 0; j < NCTX; ++j) {
        const float4 v = ((const float4*)(Wg + (size_t)xb[2 + j] * VEC_DIM))[lane];
        acc.x += v.x; acc.y += v.y; acc.z += v.z; acc.w += v.w;
    }
    float* cbuf = (float*)(base + CTX_OFF);
    ((float4*)cbuf)[lane] = acc;                    // stage c into LDS (same wave)

    // ---- Sense rows now needed: wait for async copies.
    asm volatile("s_wait_asynccnt 0" ::: "memory");

    // ---- scores(8) = S(8x128) . c(128) via 4 chained v_wmma_f32_16x16x32_f16.
    // A (16x32 f16, §7.12.2): lane L<16 row=L, K = {kb..kb+7, kb+16..kb+23};
    //                         lane L>=16 row=L-16, K shifted by +8. Rows 8..15 dup.
    // B (32x16 f16): lane n<16 holds column n = c[{kb..kb+7, kb+16..kb+23}]; lanes>=16: +8.
    const float* Srow  = (const float*)(base + (size_t)(lane & 7) * ROWB);
    const int    koffA = (lane >= 16) ? 8  : 0;
    const int    koffB = (lane >= 16) ? 8  : 0;
    v8f sc = {0.f, 0.f, 0.f, 0.f, 0.f, 0.f, 0.f, 0.f};
#pragma unroll
    for (int q = 0; q < 4; ++q) {
        const int kb = q * 32;
        const float4 a0 = *(const float4*)(Srow + kb + koffA);
        const float4 a1 = *(const float4*)(Srow + kb + koffA + 4);
        const float4 a2 = *(const float4*)(Srow + kb + 16 + koffA);
        const float4 a3 = *(const float4*)(Srow + kb + 16 + koffA + 4);
        const float4 b0 = *(const float4*)(cbuf + kb + koffB);
        const float4 b1 = *(const float4*)(cbuf + kb + koffB + 4);
        const float4 b2 = *(const float4*)(cbuf + kb + 16 + koffB);
        const float4 b3 = *(const float4*)(cbuf + kb + 16 + koffB + 4);
        v16h A, B;
        A[0]=(_Float16)a0.x; A[1]=(_Float16)a0.y; A[2]=(_Float16)a0.z; A[3]=(_Float16)a0.w;
        A[4]=(_Float16)a1.x; A[5]=(_Float16)a1.y; A[6]=(_Float16)a1.z; A[7]=(_Float16)a1.w;
        A[8]=(_Float16)a2.x; A[9]=(_Float16)a2.y; A[10]=(_Float16)a2.z; A[11]=(_Float16)a2.w;
        A[12]=(_Float16)a3.x; A[13]=(_Float16)a3.y; A[14]=(_Float16)a3.z; A[15]=(_Float16)a3.w;
        B[0]=(_Float16)b0.x; B[1]=(_Float16)b0.y; B[2]=(_Float16)b0.z; B[3]=(_Float16)b0.w;
        B[4]=(_Float16)b1.x; B[5]=(_Float16)b1.y; B[6]=(_Float16)b1.z; B[7]=(_Float16)b1.w;
        B[8]=(_Float16)b2.x; B[9]=(_Float16)b2.y; B[10]=(_Float16)b2.z; B[11]=(_Float16)b2.w;
        B[12]=(_Float16)b3.x; B[13]=(_Float16)b3.y; B[14]=(_Float16)b3.z; B[15]=(_Float16)b3.w;
        sc = __builtin_amdgcn_wmma_f32_16x16x32_f16(false, A, false, B,
                                                    (short)0, sc, false, false);
    }

    // ---- Argmax (first-max wins, matches jnp.argmax). Lanes 0-15 hold all 8
    //      scores in D VGPRs 0..7 (identical columns); broadcast from lane 0.
    int best = 0;
#pragma unroll
    for (int k = 1; k < NUM_SENSES; ++k)
        if (sc[k] > sc[best]) best = k;
    best = __shfl(best, 0, 32);

    // ---- dot(chosen_sense, W_g[x1]) in f32, butterfly reduce, sigmoid.
    const float4 s = ((const float4*)(base + (size_t)best * ROWB))[lane];
    float d = s.x * t.x + s.y * t.y + s.z * t.z + s.w * t.w;
#pragma unroll
    for (int off = 16; off >= 1; off >>= 1)
        d += __shfl_xor(d, off, 32);

    if (lane == 0)
        out[wid] = 1.0f / (1.0f + __expf(-d));
}

extern "C" void kernel_launch(void* const* d_in, const int* in_sizes, int n_in,
                              void* d_out, int out_size, void* d_ws, size_t ws_size,
                              hipStream_t stream) {
    (void)n_in; (void)out_size; (void)d_ws; (void)ws_size;
    const int*   xidx = (const int*)d_in[0];
    const float* Wg   = (const float*)d_in[1];
    const float* Ws   = (const float*)d_in[2];
    float*       out  = (float*)d_out;
    const int nrows   = in_sizes[0] / NIDX;
    const int nblocks = (nrows + WAVES_PER_BLOCK - 1) / WAVES_PER_BLOCK;
    hipLaunchKernelGGL(SenseEmbedding_kernel, dim3(nblocks), dim3(THREADS), 0, stream,
                       xidx, Wg, Ws, out, nrows);
}
